// FastNormalizedCrossCorrelation_64321430225016
// MI455X (gfx1250) — matrix operand order repaired and checked by reading, not compiled, not run on hardware
//
#include <hip/hip_runtime.h>
#include <stdint.h>

typedef __attribute__((ext_vector_type(16))) __bf16 v16bf;
typedef __attribute__((ext_vector_type(8)))  float  v8f;
typedef __attribute__((ext_vector_type(4)))  unsigned int v4u;
typedef __attribute__((ext_vector_type(4)))  int v4i;
typedef __attribute__((ext_vector_type(8)))  int v8i;

#define BCN    48          // B*C
#define HN     512
#define WN     512
#define PADL   32          // halo on all sides (left/top); extra right slack below
#define PHH    576         // padded height  = 512 + 64
#define PWW    592         // padded width   = 512 + 80 (room for +1 shifted reads)
#define RTILE  64          // output rows per workgroup
#define RCROWS 79          // RTILE + 15 rowcorr rows per phase
#define RCSTR  276         // 16*17 + 4  (bank-spread padded row stride, words)
#define STROWS 95          // staged image rows per workgroup
#define NTOT   ((size_t)BCN * HN * WN)

// ---------------------------------------------------------------------------
// TDM: async 2-D tile load (95 rows x 64 bf16) from padded global into LDS.
// 6-arg toolchain form: (g0 u32x4, g1 i32x8, g2 i32x4, g3 i32x4, i32x8, cpol)
// ---------------------------------------------------------------------------
__device__ inline void tdm2d(unsigned ldsOff, const void* gp) {
#if defined(__AMDGCN__) && __has_builtin(__builtin_amdgcn_tensor_load_to_lds)
  unsigned long long ga = (unsigned long long)(uintptr_t)gp;
  v4u g0;
  g0[0] = 1u;                                              // count=1, user desc
  g0[1] = ldsOff;                                          // LDS byte address
  g0[2] = (unsigned)ga;                                    // global addr [31:0]
  g0[3] = ((unsigned)(ga >> 32) & 0x1FFFFFFu) | (2u << 30); // addr[56:32] | type=2
  v8i g1;
  g1[0] = (int)(1u << 16);          // workgroup_mask=0, data_size=1 (2 bytes)
  g1[1] = (int)((unsigned)PWW << 16); // tensor_dim0 = PWW (row length)
  g1[2] = (int)(4096u << 16);       // tensor_dim1 = 4096 (>= rows, no clip)
  g1[3] = (int)(64u << 16);         // tile_dim0 = 64 elements
  g1[4] = 95;                       // tile_dim1 = 95 rows, tile_dim2 = 0
  g1[5] = PWW;                      // tensor_dim0_stride (elements)
  g1[6] = 0;
  g1[7] = 0;
  v4i z4 = {0, 0, 0, 0};
  v8i z8 = {0, 0, 0, 0, 0, 0, 0, 0};
  __builtin_amdgcn_tensor_load_to_lds(g0, g1, z4, z4, z8, 0);
#else
  (void)ldsOff; (void)gp;
#endif
}

// ---------------------------------------------------------------------------
// Pass 1/3: horizontal 32-tap box sum (optionally of squared input)
// ---------------------------------------------------------------------------
template <bool SQ>
__global__ void hbox_kernel(const float* __restrict__ in, float* __restrict__ out,
                            int total) {
  int idx = blockIdx.x * blockDim.x + threadIdx.x;
  if (idx >= total) return;
  int x = idx & (WN - 1);
  int rowbase = idx - x;
  float s = 0.f;
#pragma unroll
  for (int d = -15; d <= 16; ++d) {
    int xx = x + d;
    if (xx >= 0 && xx < WN) {
      float v = in[rowbase + xx];
      s += SQ ? v * v : v;
    }
  }
  out[idx] = s;
}

// ---------------------------------------------------------------------------
// Pass 2: vertical box (running sum) -> im_c = im - mean
// ---------------------------------------------------------------------------
__global__ void vbox_imc_kernel(const float* __restrict__ hs,
                                const float* __restrict__ im,
                                float* __restrict__ imc) {
  int col = blockIdx.x * blockDim.x + threadIdx.x;
  if (col >= BCN * WN) return;
  int x = col & (WN - 1);
  int bc = col >> 9;
  const float* h = hs + (size_t)bc * HN * WN + x;
  const float* ip = im + (size_t)bc * HN * WN + x;
  float* op = imc + (size_t)bc * HN * WN + x;
  float s = 0.f;
  for (int y = 0; y <= 16; ++y) s += h[(size_t)y * WN];
  for (int y = 0; y < HN; ++y) {
    op[(size_t)y * WN] = ip[(size_t)y * WN] - s * (1.0f / 1024.0f);
    int add = y + 17, sub = y - 15;
    if (add < HN) s += h[(size_t)add * WN];
    if (sub >= 0) s -= h[(size_t)sub * WN];
  }
}

// ---------------------------------------------------------------------------
// Pass 4: vertical box of im_c^2 -> energy; a = im_c/energy split to bf16
// hi/lo, written into halo-padded (PHH x PWW) arrays (halo pre-zeroed).
// ---------------------------------------------------------------------------
__global__ void vbox_energy_kernel(const float* __restrict__ hs2,
                                   const float* __restrict__ imc,
                                   __bf16* __restrict__ aHi,
                                   __bf16* __restrict__ aLo) {
  int col = blockIdx.x * blockDim.x + threadIdx.x;
  if (col >= BCN * WN) return;
  int x = col & (WN - 1);
  int bc = col >> 9;
  const float* h = hs2 + (size_t)bc * HN * WN + x;
  const float* ic = imc + (size_t)bc * HN * WN + x;
  float s = 0.f;
  for (int y = 0; y <= 16; ++y) s += h[(size_t)y * WN];
  for (int y = 0; y < HN; ++y) {
    float en = sqrtf(fmaxf(s, 0.f));
    float v = ic[(size_t)y * WN];
    float a = (en > 0.f) ? (v / en) : 0.f;
    __bf16 hi = (__bf16)a;
    __bf16 lo = (__bf16)(a - (float)hi);
    size_t p = ((size_t)bc * PHH + (size_t)(y + PADL)) * PWW + (size_t)(x + PADL);
    aHi[p] = hi;
    aLo[p] = lo;
    int add = y + 17, sub = y - 15;
    if (add < HN) s += h[(size_t)add * WN];
    if (sub >= 0) s -= h[(size_t)sub * WN];
  }
}

// ---------------------------------------------------------------------------
// Template: tn = (t - mean) / ||t - mean||, split into bf16 hi/lo.
// ---------------------------------------------------------------------------
__global__ void prep_template_kernel(const float* __restrict__ t,
                                     __bf16* __restrict__ tHi,
                                     __bf16* __restrict__ tLo) {
  __shared__ float red[256];
  __shared__ float sMean, sInv;
  int bc = blockIdx.x;
  int tid = threadIdx.x;
  const float* tp = t + (size_t)bc * 1024;
  float s = 0.f;
  for (int i = tid; i < 1024; i += 256) s += tp[i];
  red[tid] = s;
  __syncthreads();
  for (int w = 128; w > 0; w >>= 1) {
    if (tid < w) red[tid] += red[tid + w];
    __syncthreads();
  }
  if (tid == 0) sMean = red[0] * (1.f / 1024.f);
  __syncthreads();
  float mean = sMean;
  float s2 = 0.f;
  for (int i = tid; i < 1024; i += 256) {
    float c = tp[i] - mean;
    s2 += c * c;
  }
  red[tid] = s2;
  __syncthreads();
  for (int w = 128; w > 0; w >>= 1) {
    if (tid < w) red[tid] += red[tid + w];
    __syncthreads();
  }
  if (tid == 0) sInv = (red[0] > 0.f) ? (1.f / sqrtf(red[0])) : 0.f;
  __syncthreads();
  float inv = sInv;
  for (int i = tid; i < 1024; i += 256) {
    float v = (tp[i] - mean) * inv;
    __bf16 hi = (__bf16)v;
    tHi[(size_t)bc * 1024 + i] = hi;
    tLo[(size_t)bc * 1024 + i] = (__bf16)(v - (float)hi);
  }
}

// ---------------------------------------------------------------------------
// Main NCC kernel: per WG a 64x16 output tile.
//  Stage 1 (x2 phases over template-row halves): rowcorr via
//  v_wmma_f32_16x16x32_bf16, A = template rows (16x32), B = shifted image row
//  (32x16), 3 accumulated WMMAs for the bf16 hi/lo split. Stage 2: diagonal
//  LDS sum rowcorr -> out.
// ---------------------------------------------------------------------------
__global__ __launch_bounds__(256) void ncc_main(const __bf16* __restrict__ aHi,
                                                const __bf16* __restrict__ aLo,
                                                const __bf16* __restrict__ tHi,
                                                const __bf16* __restrict__ tLo,
                                                float* __restrict__ out) {
  __shared__ float rc[RCROWS * RCSTR];        // 87,216 B rowcorr scratch
  __shared__ unsigned st[4][STROWS * 32];     // 48,640 B: HiEven,HiOdd,LoEven,LoOdd
  __shared__ __bf16 tpl[2][1024];             //  4,096 B template hi/lo

  const int x0 = blockIdx.x << 4;
  const int y0 = blockIdx.y * RTILE;
  const int bc = blockIdx.z;
  const int tid = (int)threadIdx.x;
  const int lane = tid & 31;
  const int wv = tid >> 5;
  const int ln = lane & 15;   // N index / A-matrix M index
  const int lh = lane >> 4;   // lane half (K-range selector)

  for (int i = tid; i < 1024; i += 256) {
    tpl[0][i] = tHi[(size_t)bc * 1024 + i];
    tpl[1][i] = tLo[(size_t)bc * 1024 + i];
  }

  // staged tile origin: original (y0-15, x0-16) -> padded (+32,+32)
  const size_t gbase =
      ((size_t)bc * PHH + (size_t)(y0 + 17)) * PWW + (size_t)(x0 + 16);

#if defined(__AMDGCN__) && __has_builtin(__builtin_amdgcn_tensor_load_to_lds)
  if (tid == 0) {
    tdm2d((unsigned)(uintptr_t)&st[0][0], aHi + gbase);
    tdm2d((unsigned)(uintptr_t)&st[1][0], aHi + gbase + 1);
    tdm2d((unsigned)(uintptr_t)&st[2][0], aLo + gbase);
    tdm2d((unsigned)(uintptr_t)&st[3][0], aLo + gbase + 1);
    __builtin_amdgcn_s_wait_tensorcnt(0);
  }
#else
  for (int idx = tid; idx < STROWS * 32; idx += 256) {
    int r = idx >> 5, w = idx & 31;
    const unsigned* ph = (const unsigned*)(aHi + gbase + (size_t)r * PWW);
    const unsigned* pl = (const unsigned*)(aLo + gbase + (size_t)r * PWW);
    unsigned a0 = ph[w], a1 = ph[w + 1];
    unsigned b0 = pl[w], b1 = pl[w + 1];
    st[0][idx] = a0;
    st[1][idx] = (a0 >> 16) | (a1 << 16);
    st[2][idx] = b0;
    st[3][idx] = (b0 >> 16) | (b1 << 16);
  }
#endif
  __syncthreads();

  // Loop-invariant A matrices: template rows, bf16 hi/lo, two 16-row tiles.
  v16bf Ah[2], Al[2];
#pragma unroll
  for (int T = 0; T < 2; ++T) {
    const int rowb = (ln + 16 * T) * 32;
#pragma unroll
    for (int i = 0; i < 16; ++i) {
      const int l = 16 * lh + i; // A VGPR layout
      Ah[T][i] = tpl[0][rowb + l];
      Al[T][i] = tpl[1][rowb + l];
    }
  }

  // B fragment source: 16 consecutive bf16 from staged window, start col
  // cb = ln + 16*lh + 1; pick even/odd staged copy for aligned b32 loads.
  const int cb = ln + 16 * lh + 1;
  const int par = cb & 1;
  const int wb = (cb - par) >> 1;

  float acc[4] = {0.f, 0.f, 0.f, 0.f};
  const int nn = tid & 15;
  const int jb = tid >> 4;

#pragma unroll
  for (int p = 0; p < 2; ++p) {
    // ---- stage 1: rowcorr for template rows [16p, 16p+16) ----
    for (int rl = wv; rl < RCROWS; rl += 8) {
      const int sb = (rl + 16 * p) * 32 + wb;
      union { v16bf v; unsigned u[8]; } bh, bl;
#pragma unroll
      for (int i = 0; i < 8; ++i) {
        bh.u[i] = st[par][sb + i];
        bl.u[i] = st[2 + par][sb + i];
      }
      v8f d = {};
      d = __builtin_amdgcn_wmma_f32_16x16x32_bf16(false, Ah[p], false, bh.v,
                                                  (short)0, d, false, false);
      d = __builtin_amdgcn_wmma_f32_16x16x32_bf16(false, Al[p], false, bh.v,
                                                  (short)0, d, false, false);
      d = __builtin_amdgcn_wmma_f32_16x16x32_bf16(false, Ah[p], false, bl.v,
                                                  (short)0, d, false, false);
      const int rb = rl * RCSTR + ln;
#pragma unroll
      for (int j = 0; j < 8; ++j)
        rc[rb + (j + 8 * lh) * 17] = d[j];
    }
    __syncthreads();
    // ---- stage 2: out[y0+j, x0+nn] += sum_k rowcorr[j+k][k] ----
#pragma unroll
    for (int q = 0; q < 4; ++q) {
      const int j = jb + 16 * q;
      float s = 0.f;
#pragma unroll
      for (int k = 0; k < 16; ++k)
        s += rc[(j + k) * RCSTR + k * 17 + nn];
      acc[q] += s;
    }
    __syncthreads();
  }

#pragma unroll
  for (int q = 0; q < 4; ++q)
    out[((size_t)bc * HN + (size_t)(y0 + jb + 16 * q)) * WN + (x0 + nn)] =
        acc[q];
}

// ---------------------------------------------------------------------------
extern "C" void kernel_launch(void* const* d_in, const int* in_sizes, int n_in,
                              void* d_out, int out_size, void* d_ws,
                              size_t ws_size, hipStream_t stream) {
  const float* im = (const float*)d_in[0];
  const float* tpl = (const float*)d_in[1];
  float* out = (float*)d_out;
  char* ws = (char*)d_ws;

  const size_t nPad = (size_t)BCN * PHH * PWW;
  float* tmp = (float*)ws;                              // 48*512*512*4
  float* imc = (float*)(ws + NTOT * 4);                 // 48*512*512*4
  __bf16* aHi = (__bf16*)(ws + NTOT * 8);               // padded bf16
  __bf16* aLo = aHi + nPad;
  __bf16* tHi = aLo + nPad;
  __bf16* tLo = tHi + (size_t)BCN * 1024;

  // zero halos (and interiors, rewritten below) of padded a arrays
  (void)hipMemsetAsync(aHi, 0, nPad * 2 * 2, stream);

  const int tot = (int)NTOT;
  hbox_kernel<false><<<(tot + 255) / 256, 256, 0, stream>>>(im, tmp, tot);
  vbox_imc_kernel<<<(BCN * WN + 255) / 256, 256, 0, stream>>>(tmp, im, imc);
  hbox_kernel<true><<<(tot + 255) / 256, 256, 0, stream>>>(imc, tmp, tot);
  vbox_energy_kernel<<<(BCN * WN + 255) / 256, 256, 0, stream>>>(tmp, imc, aHi,
                                                                 aLo);
  prep_template_kernel<<<BCN, 256, 0, stream>>>(tpl, tHi, tLo);

  dim3 grid(WN / 16, HN / RTILE, BCN);
  ncc_main<<<grid, 256, 0, stream>>>(aHi, aLo, tHi, tLo, out);
}
